// GNNEncoder_21638045237394
// MI455X (gfx1250) — compile-verified, matrix operand order checked by reading, not run on hardware
//
#include <hip/hip_runtime.h>
#include <stdint.h>

#define NODES   50000
#define EDGES   800000
#define DIM     300
#define HID     600
#define LAYERS  5
#define KP1     320    // padded K for GEMM1 A (agg)
#define KP2     640    // padded K for GEMM2 A (hid)
#define BN_EPS  1e-5f

typedef __attribute__((ext_vector_type(16))) __bf16 v16bf;
typedef __attribute__((ext_vector_type(8)))  float  v8f;
typedef int v4i_vs __attribute__((vector_size(16)));   // matches builtin param spelling

union FragAB { v16bf v; uint4 q[2]; };

__device__ __forceinline__ unsigned short f2bf(float f) {
  unsigned u = __builtin_bit_cast(unsigned, f);
  unsigned r = u + 0x7FFFu + ((u >> 16) & 1u);   // round-to-nearest-even
  return (unsigned short)(r >> 16);
}

// ---------------- CDNA5 async global->LDS path (guarded) ----------------
#if __has_builtin(__builtin_amdgcn_global_load_async_to_lds_b128)
#define HAVE_ASYNC 1
#define ASYNC_LD_B128(GP, LP)                                              \
  __builtin_amdgcn_global_load_async_to_lds_b128(                          \
      (__attribute__((address_space(1))) v4i_vs*)(GP),                     \
      (__attribute__((address_space(3))) v4i_vs*)(LP), 0, 0)
#else
#define HAVE_ASYNC 0
#endif

__device__ __forceinline__ void wait_async0() {
#if __has_builtin(__builtin_amdgcn_s_wait_asynccnt)
  __builtin_amdgcn_s_wait_asynccnt(0);
#else
  asm volatile("s_wait_asynccnt 0" ::: "memory");
#endif
}

// ---------------- utility: zero a buffer in 16B chunks ----------------
__global__ void k_zero_u4(uint4* __restrict__ p, long n) {
  long i = (long)blockIdx.x * blockDim.x + threadIdx.x;
  if (i < n) { uint4 z = {0u,0u,0u,0u}; p[i] = z; }
}

// ---------------- node embedding: h = x_emb[x0] + x_emb[x1] ----------------
__global__ void k_node_emb(const int* __restrict__ x, const float* __restrict__ xe,
                           float* __restrict__ h) {
  int i = blockIdx.x * blockDim.x + threadIdx.x;
  if (i >= NODES * DIM) return;
  int n = i / DIM, d = i - n * DIM;
  int i0 = x[2*n], i1 = x[2*n+1];
  h[i] = xe[(size_t)i0*DIM + d] + xe[(size_t)i1*DIM + d];
}

// ---------------- scatter: agg[dst] += h[src] + ee[a0] + ee[a1] ----------------
// one wave (32 lanes) per (edge or self-loop); float4 row traffic
__global__ __launch_bounds__(256) void k_scatter(
    const int* __restrict__ eidx, const int* __restrict__ eattr,
    const float* __restrict__ ee_tab,      // [6][300] for this layer
    const float* __restrict__ h, float* __restrict__ agg) {
  int gid  = blockIdx.x * blockDim.x + threadIdx.x;
  int wid  = gid >> 5;
  int lane = threadIdx.x & 31;
  const int total = EDGES + NODES;
  if (wid >= total) return;
  int src, dst, a0, a1;
  if (wid < EDGES) {
    src = eidx[wid]; dst = eidx[EDGES + wid];
    a0 = eattr[2*wid]; a1 = eattr[2*wid + 1];
  } else {
    int nn = wid - EDGES; src = nn; dst = nn; a0 = 4; a1 = 0;   // self loop
  }
  const float4* e0 = (const float4*)(ee_tab + (size_t)a0 * DIM);
  const float4* e1 = (const float4*)(ee_tab + (size_t)a1 * DIM);
  const float4* hs = (const float4*)(h + (size_t)src * DIM);
  float*        ad = agg + (size_t)dst * DIM;
  // DIM = 300 floats = 75 float4
  for (int q = lane; q < DIM / 4; q += 32) {
    float4 a = hs[q], b = e0[q], c = e1[q];
    float* p = ad + q * 4;
    __hip_atomic_fetch_add(p + 0, a.x + b.x + c.x, __ATOMIC_RELAXED, __HIP_MEMORY_SCOPE_AGENT);
    __hip_atomic_fetch_add(p + 1, a.y + b.y + c.y, __ATOMIC_RELAXED, __HIP_MEMORY_SCOPE_AGENT);
    __hip_atomic_fetch_add(p + 2, a.z + b.z + c.z, __ATOMIC_RELAXED, __HIP_MEMORY_SCOPE_AGENT);
    __hip_atomic_fetch_add(p + 3, a.w + b.w + c.w, __ATOMIC_RELAXED, __HIP_MEMORY_SCOPE_AGENT);
  }
}

// ---------------- f32 -> bf16 with K padding (activations) ----------------
__global__ void k_f32_to_bf16_pad(const float* __restrict__ src,
                                  unsigned short* __restrict__ dst,
                                  int rows, int cols, int ldDst) {
  int i = blockIdx.x * blockDim.x + threadIdx.x;
  if (i >= rows * cols) return;
  int r = i / cols, c = i - r * cols;
  dst[(size_t)r * ldDst + c] = f2bf(src[i]);
}

// ---------------- weight transpose+convert: WT[n][k] = bf16(W[k][n]) ----------------
__global__ void k_w_to_bf16_t(const float* __restrict__ W,
                              unsigned short* __restrict__ WT,
                              int Kreal, int N, int Kpad) {
  int i = blockIdx.x * blockDim.x + threadIdx.x;
  if (i >= N * Kpad) return;
  int n = i / Kpad, k = i - n * Kpad;
  float v = (k < Kreal) ? W[(size_t)k * N + n] : 0.f;
  WT[i] = f2bf(v);
}

// ---------------- bf16 WMMA GEMM: out = act( A[MxK]*B[KxN] + bias ) ----------------
// A: bf16 [M x Kpad], BT: bf16 [N x Kpad] (B transposed). Block=256 (8 waves),
// tile M=128, N=64, Kstep=32; each wave computes 16x64 (4 WMMAs per step).
// OOB tile rows read harmless garbage (epilogue stores are guarded).
#define LDA_S 40
#define LDB_S 40
#if HAVE_ASYNC
#define NBUF 2
#else
#define NBUF 1
#endif
__global__ __launch_bounds__(256) void k_gemm_bf16(
    const unsigned short* __restrict__ A, int lda,
    const unsigned short* __restrict__ BT, int ldbt,
    const float* __restrict__ bias,
    int M, int Ncols, int kSteps, int mode,
    unsigned short* __restrict__ outB, int ldOutB,
    float* __restrict__ outF, int ldOutF) {
  __shared__ __align__(16) unsigned short sA[NBUF][128 * LDA_S];
  __shared__ __align__(16) unsigned short sB[NBUF][64 * LDB_S];
  const int t    = threadIdx.x;
  const int wv   = t >> 5;
  const int lane = t & 31;
  const int m    = lane & 15;
  const int hi   = lane >> 4;
  const int mBase = blockIdx.x * 128;
  const int nBase = blockIdx.y * 64;
  const int ra = t >> 2, sg = (t & 3) * 8;      // tile row / 8-element segment

  v8f acc[4];
  for (int n = 0; n < 4; ++n)
    for (int i = 0; i < 8; ++i) acc[n][i] = 0.f;

#if HAVE_ASYNC
  // ---- double-buffered async pipeline ----
  #define ISSUE_TILES(KS, B_) do {                                              \
    int kb_ = (KS) * 32;                                                        \
    ASYNC_LD_B128(A  + (size_t)(mBase + ra)      * lda  + kb_ + sg,             \
                  &sA[B_][ra * LDA_S + sg]);                                    \
    ASYNC_LD_B128(A  + (size_t)(mBase + ra + 64) * lda  + kb_ + sg,             \
                  &sA[B_][(ra + 64) * LDA_S + sg]);                             \
    ASYNC_LD_B128(BT + (size_t)(nBase + ra)      * ldbt + kb_ + sg,             \
                  &sB[B_][ra * LDB_S + sg]);                                    \
  } while (0)

  int buf = 0;
  ISSUE_TILES(0, 0);
  for (int ks = 0; ks < kSteps; ++ks) {
    wait_async0();
    __syncthreads();                       // tiles for ks landed in LDS
    if (ks + 1 < kSteps) ISSUE_TILES(ks + 1, buf ^ 1);
    FragAB a;
    const unsigned short* ap = &sA[buf][(wv * 16 + m) * LDA_S + 8 * hi];
    a.q[0] = *(const uint4*)(ap);
    a.q[1] = *(const uint4*)(ap + 16);
    #pragma unroll
    for (int nt = 0; nt < 4; ++nt) {
      FragAB b;
      const unsigned short* bp = &sB[buf][(nt * 16 + m) * LDB_S + 8 * hi];
      b.q[0] = *(const uint4*)(bp);
      b.q[1] = *(const uint4*)(bp + 16);
      acc[nt] = __builtin_amdgcn_wmma_f32_16x16x32_bf16(
          false, a.v, false, b.v, (short)0, acc[nt], false, false);
    }
    __syncthreads();                       // done reading buf before it is refilled
    buf ^= 1;
  }
  #undef ISSUE_TILES
#else
  // ---- fallback: synchronous vectorized copies, single buffer ----
  for (int ks = 0; ks < kSteps; ++ks) {
    int kb = ks * 32;
    uint4 qa0 = *(const uint4*)(A  + (size_t)(mBase + ra)      * lda  + kb + sg);
    uint4 qa1 = *(const uint4*)(A  + (size_t)(mBase + ra + 64) * lda  + kb + sg);
    uint4 qb  = *(const uint4*)(BT + (size_t)(nBase + ra)      * ldbt + kb + sg);
    __syncthreads();                       // prior compute done reading LDS
    *(uint4*)(&sA[0][ra * LDA_S + sg])        = qa0;
    *(uint4*)(&sA[0][(ra + 64) * LDA_S + sg]) = qa1;
    *(uint4*)(&sB[0][ra * LDB_S + sg])        = qb;
    __syncthreads();
    FragAB a;
    const unsigned short* ap = &sA[0][(wv * 16 + m) * LDA_S + 8 * hi];
    a.q[0] = *(const uint4*)(ap);
    a.q[1] = *(const uint4*)(ap + 16);
    #pragma unroll
    for (int nt = 0; nt < 4; ++nt) {
      FragAB b;
      const unsigned short* bp = &sB[0][(nt * 16 + m) * LDB_S + 8 * hi];
      b.q[0] = *(const uint4*)(bp);
      b.q[1] = *(const uint4*)(bp + 16);
      acc[nt] = __builtin_amdgcn_wmma_f32_16x16x32_bf16(
          false, a.v, false, b.v, (short)0, acc[nt], false, false);
    }
  }
#endif

  // ---- epilogue: C/D layout: row = i + 8*hi, col = lane&15 ----
  #pragma unroll
  for (int nt = 0; nt < 4; ++nt) {
    int col = nBase + nt * 16 + m;
    if (col >= Ncols) continue;
    float bv = bias[col];
    #pragma unroll
    for (int i = 0; i < 8; ++i) {
      int row = mBase + wv * 16 + i + 8 * hi;
      if (row >= M) continue;
      float v = acc[nt][i] + bv;
      if (mode == 0) {
        v = fmaxf(v, 0.f);
        outB[(size_t)row * ldOutB + col] = f2bf(v);
      } else {
        outF[(size_t)row * ldOutF + col] = v;
      }
    }
  }
}

// ---------------- BatchNorm stats: one block per column ----------------
__global__ __launch_bounds__(256) void k_bn_stats(const float* __restrict__ pre,
                                                  float* __restrict__ meanA,
                                                  float* __restrict__ rstdA) {
  const int c = blockIdx.x, t = threadIdx.x;
  float s = 0.f, s2 = 0.f;
  for (int r = t; r < NODES; r += 256) {
    float v = pre[(size_t)r * DIM + c];
    s += v; s2 += v * v;
  }
  __shared__ float sh[256], sh2[256];
  sh[t] = s; sh2[t] = s2; __syncthreads();
  for (int o = 128; o > 0; o >>= 1) {
    if (t < o) { sh[t] += sh[t + o]; sh2[t] += sh2[t + o]; }
    __syncthreads();
  }
  if (t == 0) {
    float mu  = sh[0]  * (1.f / NODES);
    float var = sh2[0] * (1.f / NODES) - mu * mu;
    meanA[c] = mu;
    rstdA[c] = rsqrtf(var + BN_EPS);
  }
}

// ---------------- BatchNorm apply (+optional ReLU) ----------------
__global__ void k_bn_apply(const float* __restrict__ pre,
                           const float* __restrict__ meanA, const float* __restrict__ rstdA,
                           const float* __restrict__ gam, const float* __restrict__ bet,
                           float* __restrict__ out, int relu) {
  int i = blockIdx.x * blockDim.x + threadIdx.x;
  if (i >= NODES * DIM) return;
  int c = i % DIM;
  float v = (pre[i] - meanA[c]) * rstdA[c] * gam[c] + bet[c];
  if (relu) v = fmaxf(v, 0.f);
  out[i] = v;
}

// =====================================================================
extern "C" void kernel_launch(void* const* d_in, const int* in_sizes, int n_in,
                              void* d_out, int out_size, void* d_ws, size_t ws_size,
                              hipStream_t stream) {
  (void)in_sizes; (void)n_in; (void)out_size; (void)ws_size;
  const int*   x        = (const int*)d_in[0];
  const int*   eidx     = (const int*)d_in[1];
  const int*   eattr    = (const int*)d_in[2];
  const float* x_emb    = (const float*)d_in[3];
  const float* edge_emb = (const float*)d_in[4];
  const float* W1       = (const float*)d_in[5];
  const float* b1       = (const float*)d_in[6];
  const float* W2       = (const float*)d_in[7];
  const float* b2       = (const float*)d_in[8];
  const float* gam      = (const float*)d_in[9];
  const float* bet      = (const float*)d_in[10];
  float* outF = (float*)d_out;

  char* ws = (char*)d_ws;
  float*          hF    = (float*)(ws);                         // 60 MB
  float*          aggF  = (float*)(ws + 60000000L);             // 60 MB
  unsigned short* aggB  = (unsigned short*)(ws + 120000000L);   // 32 MB
  unsigned short* hidB  = (unsigned short*)(ws + 152000000L);   // 64 MB
  float*          meanA = (float*)(ws + 216000000L);
  float*          rstdA = (float*)(ws + 216004096L);
  unsigned short* WT1   = (unsigned short*)(ws + 216008192L);   // 600*320*2 = 384000 B
  unsigned short* WT2   = (unsigned short*)(ws + 216392192L);   // 300*640*2 = 384000 B
  // region ends ~216.78 MB (+32KB slack consumed only by harmless OOB tile reads)

  const int NE = NODES * DIM;                 // 15e6
  dim3 blk(256);

  // zero bf16 activation staging once per launch (keeps K-pad columns == 0)
  {
    long n1 = (long)NODES * KP1 * 2 / 16;
    long n2 = (long)NODES * KP2 * 2 / 16;
    k_zero_u4<<<dim3((unsigned)((n1 + 255) / 256)), blk, 0, stream>>>((uint4*)aggB, n1);
    k_zero_u4<<<dim3((unsigned)((n2 + 255) / 256)), blk, 0, stream>>>((uint4*)hidB, n2);
  }

  // h0 = x_emb[x0] + x_emb[x1]
  k_node_emb<<<dim3((NE + 255) / 256), blk, 0, stream>>>(x, x_emb, hF);

  for (int layer = 0; layer < LAYERS; ++layer) {
    // agg = 0
    {
      long nz = (long)NODES * DIM * 4 / 16;
      k_zero_u4<<<dim3((unsigned)((nz + 255) / 256)), blk, 0, stream>>>((uint4*)aggF, nz);
    }
    // scatter-add messages (edges + self-loops)
    {
      int waves  = EDGES + NODES;
      int blocks = (waves + 7) / 8;           // 8 waves / 256-thread block
      k_scatter<<<dim3(blocks), blk, 0, stream>>>(
          eidx, eattr, edge_emb + (size_t)layer * 6 * DIM, hF, aggF);
    }
    // stage this layer's weights as transposed bf16 (tiny, L2-resident)
    k_w_to_bf16_t<<<dim3((HID * KP1 + 255) / 256), blk, 0, stream>>>(
        W1 + (size_t)layer * DIM * HID, WT1, DIM, HID, KP1);
    k_w_to_bf16_t<<<dim3((DIM * KP2 + 255) / 256), blk, 0, stream>>>(
        W2 + (size_t)layer * HID * DIM, WT2, HID, DIM, KP2);
    // agg f32 -> bf16 (K padded to 320)
    k_f32_to_bf16_pad<<<dim3((NE + 255) / 256), blk, 0, stream>>>(aggF, aggB, NODES, DIM, KP1);

    // GEMM1: hid = relu(agg @ W1 + b1)   [50000x300]@[300x600] -> bf16 hidB
    {
      dim3 grid((NODES + 127) / 128, (HID + 63) / 64);
      k_gemm_bf16<<<grid, blk, 0, stream>>>(
          aggB, KP1, WT1, KP1,
          b1 + (size_t)layer * HID,
          NODES, HID, KP1 / 32, /*mode=*/0,
          hidB, KP2, nullptr, 0);
    }
    // GEMM2: pre = hid @ W2 + b2         [50000x600]@[600x300] -> f32 hF
    {
      dim3 grid((NODES + 127) / 128, (DIM + 63) / 64);
      k_gemm_bf16<<<grid, blk, 0, stream>>>(
          hidB, KP2, WT2, KP2,
          b2 + (size_t)layer * DIM,
          NODES, DIM, KP2 / 32, /*mode=*/1,
          nullptr, 0, hF, DIM);
    }
    // BatchNorm (training-mode batch stats) + ReLU (except last layer)
    k_bn_stats<<<dim3(DIM), blk, 0, stream>>>(hF, meanA, rstdA);
    float* dst  = (layer == LAYERS - 1) ? outF : hF;
    int    relu = (layer != LAYERS - 1) ? 1 : 0;
    k_bn_apply<<<dim3((NE + 255) / 256), blk, 0, stream>>>(
        hF, meanA, rstdA,
        gam + (size_t)layer * DIM, bet + (size_t)layer * DIM, dst, relu);
  }
}
